// SurfaceTransformer_83425444757841
// MI455X (gfx1250) — matrix operand order repaired and checked by reading, not compiled, not run on hardware
//
#include <hip/hip_runtime.h>
#include <hip/hip_bf16.h>
#include <math.h>

// ---------------------------------------------------------------------------
// SurfaceTransformer on MI455X (gfx1250, wave32, WMMA + Tensor Data Mover).
//
// 0.66 TFLOP of attention MLP -> v_wmma_f32_16x16x32_f16 (f32 accumulate).
// GEMM2 + softmax + neighbor-reduction fused (sim never hits HBM).
// Tile staging global->LDS uses TENSOR_LOAD_TO_LDS (TDM) with double
// buffering: wave0 issues descriptors, s_wait_tensorcnt, barrier; the DMA
// engine fills buffer b^1 while 8 waves run WMMA out of buffer b.
// Weights are pre-packed in global memory into the exact LDS image the
// B fragment needs, so TDM is a raw 2D tile copy; TDM pad_interval/amount
// reproduce the bank-conflict-avoiding strides (34 halves / 65 dwords).
// ---------------------------------------------------------------------------

typedef __attribute__((ext_vector_type(16))) _Float16 v16h;
typedef __attribute__((ext_vector_type(8)))  float    v8f;

#define BATCH 4
#define NPTS  1225
#define DIM_  256
#define KNN   32
#define ROWS  (BATCH * NPTS)        // 4900
#define TOK   (BATCH * NPTS * KNN)  // 156800

#define AS_STRIDE 34                 // halves per A-tile row in LDS (32 + pad)
#define WS_STRIDE 65                 // u32 per W-tile row in LDS (64 + pad)

union H2 { _Float16 h[2]; unsigned u; };
union AF { unsigned u[8]; v16h v; };

__device__ __forceinline__ float gelu_erf(float x) {
  return 0.5f * x * (1.0f + erff(x * 0.70710678118654752f));
}

// ---------------------------------------------------------------------------
// Tensor Data Mover support (probe-guarded; sync staging fallback otherwise)
// ---------------------------------------------------------------------------
#if defined(__gfx1250__) && __has_builtin(__builtin_amdgcn_tensor_load_to_lds) && \
    __has_builtin(__builtin_amdgcn_s_wait_tensorcnt)
#define CDNA5_TDM 1

typedef unsigned tdm_v4u __attribute__((ext_vector_type(4)));
typedef int      tdm_v4i __attribute__((ext_vector_type(4)));
typedef int      tdm_v8i __attribute__((ext_vector_type(8)));

// Generic pointers to LDS carry the LDS byte offset in their low 32 bits
// (the LDS aperture keeps the low 32 address bits as the LDS offset).
__device__ __forceinline__ unsigned lds_byte_off(const void* p) {
  return (unsigned)(unsigned long long)p;
}

// 2D tile load via TDM descriptor (D# groups per CDNA5 ISA).
//  dsz_log: 0=1B 1=2B 2=4B ; dims/strides in elements ; pad codes per spec
//  (pad_int: 3 => every 16 DW, 5 => every 64 DW ; pad_amt 0 => 1 DW).
__device__ __forceinline__ void tdm_load_2d(
    unsigned lds_off, const void* gptr, unsigned dsz_log,
    unsigned td0, unsigned td1, unsigned long long stride0,
    unsigned tile0, unsigned tile1, unsigned pad_int, unsigned pad_amt) {
  unsigned long long ga = (unsigned long long)gptr;
  union { unsigned w[4]; tdm_v4u v; } g0;
  union { int w[8]; tdm_v8i v; } g1;
  union { int w[4]; tdm_v4i v; } gz;
  // group0: count=1 | lds_addr | global_addr[56:0] | type=2 (bits 127:126)
  g0.w[0] = 1u;
  g0.w[1] = lds_off;
  g0.w[2] = (unsigned)ga;
  g0.w[3] = (unsigned)((ga >> 32) & 0x01ffffffull) | 0x80000000u;
  // group1: data_size<<16 | pad_enable<<20 | pad_interval<<22 | pad_amount<<25
  g1.w[0] = (int)((dsz_log << 16) | (1u << 20) | (pad_int << 22) | (pad_amt << 25));
  g1.w[1] = (int)((td0 & 0xffffu) << 16);                 // tensor_dim0[15:0]
  g1.w[2] = (int)((td0 >> 16) | ((td1 & 0xffffu) << 16)); // td0[31:16]|td1[15:0]
  g1.w[3] = (int)((td1 >> 16) | (tile0 << 16));           // td1[31:16]|tile_dim0
  g1.w[4] = (int)(tile1 & 0xffffu);                       // tile_dim1 (tile_dim2=0)
  g1.w[5] = (int)(unsigned)stride0;                       // dim0_stride[31:0]
  g1.w[6] = (int)(unsigned)((stride0 >> 32) & 0xffffull); // dim0_stride[47:32]
  g1.w[7] = 0;
  gz.w[0] = gz.w[1] = gz.w[2] = gz.w[3] = 0;
#if __clang_major__ >= 23
  union { int w[8]; tdm_v8i v; } gz8;
  for (int i = 0; i < 8; ++i) gz8.w[i] = 0;
  __builtin_amdgcn_tensor_load_to_lds(g0.v, g1.v, gz.v, gz.v, gz8.v, 0);
#else
  __builtin_amdgcn_tensor_load_to_lds(g0.v, g1.v, gz.v, gz.v, 0);
#endif
}
#else
#define CDNA5_TDM 0
#endif

// ---------------------------------------------------------------------------
// Weight pre-pack: f32 W[Ktot,N] -> u32 pairs Wp[Ktot/2][N], where
// Wp[p][n] = (f16(W[2p][n]), f16(W[2p+1][n])).  This is exactly the LDS
// B-fragment image, so tile staging becomes a raw 2D copy (TDM-able).
// Layer boundaries never straddle a pair (K even).
// ---------------------------------------------------------------------------
__global__ void pack_w_k(const float* __restrict__ src, unsigned* __restrict__ dst,
                         int Ktot, int N) {
  int i = blockIdx.x * 256 + threadIdx.x;
  int total = (Ktot >> 1) * N;
  if (i >= total) return;
  int p = i / N, n = i - p * N;
  H2 t;
  t.h[0] = (_Float16)src[(size_t)(2 * p) * N + n];
  t.h[1] = (_Float16)src[(size_t)(2 * p + 1) * N + n];
  dst[i] = t.u;
}

// ---------------------------------------------------------------------------
// Embedding: Linear(3->256) + GELU(erf) + LayerNorm.  One block per point.
// ---------------------------------------------------------------------------
__global__ __launch_bounds__(256) void embed_k(
    const float* __restrict__ xyz, const float* __restrict__ w,
    const float* __restrict__ bb, const float* __restrict__ g,
    const float* __restrict__ be, float* __restrict__ X) {
  __shared__ float red[256];
  int row = blockIdx.x, d = threadIdx.x;
  float c0 = xyz[row * 3], c1 = xyz[row * 3 + 1], c2 = xyz[row * 3 + 2];
  float u = bb[d] + c0 * w[d] + c1 * w[256 + d] + c2 * w[512 + d];
  float ge = gelu_erf(u);
  red[d] = ge; __syncthreads();
  for (int st = 128; st > 0; st >>= 1) { if (d < st) red[d] += red[d + st]; __syncthreads(); }
  float mean = red[0] * (1.0f / 256.0f);
  __syncthreads();
  float dv = ge - mean;
  red[d] = dv * dv; __syncthreads();
  for (int st = 128; st > 0; st >>= 1) { if (d < st) red[d] += red[d + st]; __syncthreads(); }
  float var = red[0] * (1.0f / 256.0f);
  X[(size_t)row * DIM_ + d] = dv * rsqrtf(var + 1e-5f) * g[d] + be[d];
}

// ---------------------------------------------------------------------------
// kNN (K=32 smallest squared distances, includes self).  One thread per
// query point; sorted candidate list in LDS ([entry][thread] layout).
// Neighbor order is irrelevant downstream (softmax over K is perm-invariant).
// ---------------------------------------------------------------------------
#define KNN_T 128
__global__ __launch_bounds__(KNN_T) void knn_k(const float* __restrict__ xyz,
                                               int* __restrict__ idx,
                                               float* __restrict__ rel) {
  __shared__ float dl[KNN * KNN_T];
  __shared__ int   il[KNN * KNN_T];
  int t = threadIdx.x;
  int p = blockIdx.x * KNN_T + t;
  if (p >= ROWS) return;
  int b = p / NPTS;
  const float* base = xyz + (size_t)b * NPTS * 3;
  float x0 = xyz[p * 3], x1 = xyz[p * 3 + 1], x2 = xyz[p * 3 + 2];
  int m = 0;
  float worst = 3.0e38f;
  for (int j = 0; j < NPTS; ++j) {
    float d0 = x0 - base[j * 3], d1 = x1 - base[j * 3 + 1], d2 = x2 - base[j * 3 + 2];
    float dd = d0 * d0 + d1 * d1 + d2 * d2;
    if (m < KNN || dd < worst) {
      int pos = (m < KNN) ? m : (KNN - 1);
      while (pos > 0 && dl[(pos - 1) * KNN_T + t] > dd) {
        dl[pos * KNN_T + t] = dl[(pos - 1) * KNN_T + t];
        il[pos * KNN_T + t] = il[(pos - 1) * KNN_T + t];
        --pos;
      }
      dl[pos * KNN_T + t] = dd;
      il[pos * KNN_T + t] = j;
      if (m < KNN) ++m;
      if (m == KNN) worst = dl[(KNN - 1) * KNN_T + t];
    }
  }
  for (int e = 0; e < KNN; ++e) {
    int j = il[e * KNN_T + t];
    idx[(size_t)p * KNN + e] = j;
    size_t ro = ((size_t)p * KNN + e) * 3;
    rel[ro + 0] = x0 - base[j * 3];
    rel[ro + 1] = x1 - base[j * 3 + 1];
    rel[ro + 2] = x2 - base[j * 3 + 2];
  }
}

// ---------------------------------------------------------------------------
// WMMA GEMM: out[M,N] = A[M,K] @ W[K,N] (+bias)(+relu), f32 accumulate.
// 256 threads = 8 wave32s; block tile 128x64; BK=32; double-buffered LDS.
// A_TDM: stage A via TDM (requires f16 A); W always TDM when available.
// ---------------------------------------------------------------------------
template<bool A_F16, bool A_TDM, bool RELU, bool BIAS, bool OUT_F16>
__global__ __launch_bounds__(256) void gemm_wmma_k(
    const void* __restrict__ Aptr, const unsigned* __restrict__ Wp,
    const float* __restrict__ bias, void* __restrict__ Optr,
    int M, int N, int K) {
  __shared__ _Float16 As[2 * 128 * AS_STRIDE];
  __shared__ unsigned Ws[2 * 16 * WS_STRIDE];
  const _Float16* Ah = (const _Float16*)Aptr;
  const float*    Af = (const float*)Aptr;
  _Float16* Oh = (_Float16*)Optr;
  float*    Of = (float*)Optr;

  int tid = threadIdx.x;
  int lane = tid & 31, w = tid >> 5;
  int li = lane & 15, hi = lane >> 4;
  int m0 = blockIdx.x * 128;
  int n0 = blockIdx.y * 64;
  const int ASZ = 128 * AS_STRIDE;   // halves per buffer
  const int WSZ = 16 * WS_STRIDE;    // u32 per buffer

  v8f acc[4] = {};

#if CDNA5_TDM
  unsigned asB = lds_byte_off(As);
  unsigned wsB = lds_byte_off(Ws);
  if (w == 0) {                      // prologue: fill buffer 0
    if (A_TDM)
      tdm_load_2d(asB, Ah + (size_t)m0 * K, 1u, (unsigned)K, (unsigned)M,
                  (unsigned long long)K, 32u, 128u, 3u, 0u);
    tdm_load_2d(wsB, Wp + (size_t)n0, 2u, (unsigned)N, (unsigned)(K >> 1),
                (unsigned long long)N, 64u, 16u, 5u, 0u);
  }
#endif

  int nk = K >> 5;
  for (int it = 0; it < nk; ++it) {
    int kk = it << 5;
    int buf = it & 1;
    __syncthreads();                 // (a) readers of buf^1 are done
#if CDNA5_TDM
    if (w == 0) {
      if (it + 1 < nk) {             // issue next tile into buf^1
        int kn = kk + 32, bn = buf ^ 1;
        if (A_TDM)
          tdm_load_2d(asB + bn * (ASZ * 2), Ah + (size_t)m0 * K + kn, 1u,
                      (unsigned)K, (unsigned)M, (unsigned long long)K,
                      32u, 128u, 3u, 0u);
        tdm_load_2d(wsB + bn * (WSZ * 4), Wp + (size_t)(kn >> 1) * N + n0, 2u,
                    (unsigned)N, (unsigned)(K >> 1), (unsigned long long)N,
                    64u, 16u, 5u, 0u);
        // in-order TENSORcnt: leaving <= (#just issued) pending => buf ready
        if (A_TDM) __builtin_amdgcn_s_wait_tensorcnt(2);
        else       __builtin_amdgcn_s_wait_tensorcnt(1);
      } else {
        __builtin_amdgcn_s_wait_tensorcnt(0);
      }
    }
    if (!A_TDM) {
      // synchronous A staging (f32 source -> f16 pairs)
      #pragma unroll
      for (int i = 0; i < 8; ++i) {
        int u = i * 256 + tid;
        int r = u >> 4, cp = u & 15;
        int grow = m0 + r;
        unsigned val = 0u;
        if (grow < M) {
          if (A_F16) {
            val = *(const unsigned*)(Ah + (size_t)grow * K + kk + cp * 2);
          } else {
            H2 tpk;
            tpk.h[0] = (_Float16)Af[(size_t)grow * K + kk + cp * 2];
            tpk.h[1] = (_Float16)Af[(size_t)grow * K + kk + cp * 2 + 1];
            val = tpk.u;
          }
        }
        *(unsigned*)&As[buf * ASZ + r * AS_STRIDE + cp * 2] = val;
      }
    }
#else
    // ---- fully synchronous fallback staging ----
    #pragma unroll
    for (int i = 0; i < 8; ++i) {
      int u = i * 256 + tid;
      int r = u >> 4, cp = u & 15;
      int grow = m0 + r;
      unsigned val = 0u;
      if (grow < M) {
        if (A_F16) {
          val = *(const unsigned*)(Ah + (size_t)grow * K + kk + cp * 2);
        } else {
          H2 tpk;
          tpk.h[0] = (_Float16)Af[(size_t)grow * K + kk + cp * 2];
          tpk.h[1] = (_Float16)Af[(size_t)grow * K + kk + cp * 2 + 1];
          val = tpk.u;
        }
      }
      *(unsigned*)&As[buf * ASZ + r * AS_STRIDE + cp * 2] = val;
    }
    #pragma unroll
    for (int i = 0; i < 4; ++i) {
      int u = i * 256 + tid;
      int pp = u >> 6, c = u & 63;
      Ws[buf * WSZ + pp * WS_STRIDE + c] = Wp[(size_t)((kk >> 1) + pp) * N + n0 + c];
    }
#endif
    __syncthreads();                 // (b) buf ready for all waves
    // ---- A fragment: lane<16 -> K {0..7,16..23}; lane>=16 -> +8 ----
    AF a;
    #pragma unroll
    for (int v = 0; v < 8; ++v) {
      int base = 2 * (v & 3) + 16 * (v >> 2) + 8 * hi;
      a.u[v] = *(const unsigned*)&As[buf * ASZ + (w * 16 + li) * AS_STRIDE + base];
    }
    #pragma unroll
    for (int ct = 0; ct < 4; ++ct) {
      AF bfr;
      #pragma unroll
      for (int v = 0; v < 8; ++v)
        bfr.u[v] = Ws[buf * WSZ + ((v & 3) + 4 * hi + 8 * (v >> 2)) * WS_STRIDE + ct * 16 + li];
      acc[ct] = __builtin_amdgcn_wmma_f32_16x16x32_f16(
          false, a.v, false, bfr.v, (short)0, acc[ct], false, false);
    }
  }

  // ---- epilogue: C layout VGPR r -> row r + 8*(lane>=16), col = lane&15 ----
  #pragma unroll
  for (int ct = 0; ct < 4; ++ct) {
    int col = n0 + ct * 16 + li;
    float bv = BIAS ? bias[col] : 0.0f;
    #pragma unroll
    for (int r = 0; r < 8; ++r) {
      int row = m0 + w * 16 + r + 8 * hi;
      if (row < M) {
        float v = acc[ct][r] + bv;
        if (RELU) v = fmaxf(v, 0.0f);
        if (OUT_F16) Oh[(size_t)row * N + col] = (_Float16)v;
        else         Of[(size_t)row * N + col] = v;
      }
    }
  }
}

// ---------------------------------------------------------------------------
// Token builder: per (b,n) block, loop k over 32 neighbors:
//   pos = ReLU(rel @ pos_w1 + b1) @ pos_w2 + b2
//   a_in = (f16)(q - k_j + pos) ; vpj = (f16)(v_j + pos)
// ---------------------------------------------------------------------------
__global__ __launch_bounds__(256) void build_tokens_k(
    const float* __restrict__ qkv, const int* __restrict__ idx,
    const float* __restrict__ rel,
    const float* __restrict__ pw1, const float* __restrict__ pb1,
    const float* __restrict__ pw2, const float* __restrict__ pb2,
    _Float16* __restrict__ ain, _Float16* __restrict__ vpj) {
  __shared__ float hh[64];
  int p = blockIdx.x;
  int b = p / NPTS;
  int d = threadIdx.x;
  float qd = qkv[(size_t)p * 768 + d];
  for (int k = 0; k < KNN; ++k) {
    __syncthreads();
    int j = idx[(size_t)p * KNN + k];
    const float* rp = rel + ((size_t)p * KNN + k) * 3;
    float r0 = rp[0], r1 = rp[1], r2 = rp[2];
    if (d < 64)
      hh[d] = fmaxf(r0 * pw1[d] + r1 * pw1[64 + d] + r2 * pw1[128 + d] + pb1[d], 0.0f);
    __syncthreads();
    float pos = pb2[d];
    #pragma unroll 8
    for (int h = 0; h < 64; ++h) pos += hh[h] * pw2[(size_t)h * 256 + d];
    size_t kvrow = (size_t)(b * NPTS + j) * 768;
    float av = qd - qkv[kvrow + 256 + d] + pos;
    float vv = qkv[kvrow + 512 + d] + pos;
    size_t o = ((size_t)p * KNN + k) * 256 + d;
    ain[o] = (_Float16)av;
    vpj[o] = (_Float16)vv;
  }
}

// ---------------------------------------------------------------------------
// Fused: sim = h @ W2 + b2 ; softmax over K=32 neighbors (block-local: one
// 128-row tile == 4 complete neighbor groups) ; x_out = sum_k attn * vpj.
// A (=h) and W2 tiles staged via TDM double buffering when available.
// ---------------------------------------------------------------------------
#define SIM_STRIDE 65
__global__ __launch_bounds__(256) void attn2_softmax_k(
    const _Float16* __restrict__ Hm, const unsigned* __restrict__ Wp,
    const float* __restrict__ bias, const _Float16* __restrict__ Vpj,
    float* __restrict__ Xout, int K) {
  const int N = 256;
  __shared__ _Float16 As[2 * 128 * AS_STRIDE];
  __shared__ unsigned Ws[2 * 16 * WS_STRIDE];
  __shared__ float simLds[128 * SIM_STRIDE];

  int tid = threadIdx.x;
  int lane = tid & 31, w = tid >> 5;
  int li = lane & 15, hi = lane >> 4;
  int m0 = blockIdx.x * 128;
  int n0 = blockIdx.y * 64;
  const int ASZ = 128 * AS_STRIDE;
  const int WSZ = 16 * WS_STRIDE;

  v8f acc[4] = {};

#if CDNA5_TDM
  unsigned asB = lds_byte_off(As);
  unsigned wsB = lds_byte_off(Ws);
  if (w == 0) {
    tdm_load_2d(asB, Hm + (size_t)m0 * K, 1u, (unsigned)K, (unsigned)TOK,
                (unsigned long long)K, 32u, 128u, 3u, 0u);
    tdm_load_2d(wsB, Wp + (size_t)n0, 2u, (unsigned)N, (unsigned)(K >> 1),
                (unsigned long long)N, 64u, 16u, 5u, 0u);
  }
#endif

  int nk = K >> 5;
  for (int it = 0; it < nk; ++it) {
    int kk = it << 5;
    int buf = it & 1;
    __syncthreads();
#if CDNA5_TDM
    if (w == 0) {
      if (it + 1 < nk) {
        int kn = kk + 32, bn = buf ^ 1;
        tdm_load_2d(asB + bn * (ASZ * 2), Hm + (size_t)m0 * K + kn, 1u,
                    (unsigned)K, (unsigned)TOK, (unsigned long long)K,
                    32u, 128u, 3u, 0u);
        tdm_load_2d(wsB + bn * (WSZ * 4), Wp + (size_t)(kn >> 1) * N + n0, 2u,
                    (unsigned)N, (unsigned)(K >> 1), (unsigned long long)N,
                    64u, 16u, 5u, 0u);
        __builtin_amdgcn_s_wait_tensorcnt(2);
      } else {
        __builtin_amdgcn_s_wait_tensorcnt(0);
      }
    }
#else
    #pragma unroll
    for (int i = 0; i < 8; ++i) {
      int u = i * 256 + tid;
      int r = u >> 4, cp = u & 15;
      *(unsigned*)&As[buf * ASZ + r * AS_STRIDE + cp * 2] =
          *(const unsigned*)(Hm + (size_t)(m0 + r) * K + kk + cp * 2);
    }
    #pragma unroll
    for (int i = 0; i < 4; ++i) {
      int u = i * 256 + tid;
      int pp = u >> 6, c = u & 63;
      Ws[buf * WSZ + pp * WS_STRIDE + c] = Wp[(size_t)((kk >> 1) + pp) * N + n0 + c];
    }
#endif
    __syncthreads();
    AF a;
    #pragma unroll
    for (int v = 0; v < 8; ++v) {
      int base = 2 * (v & 3) + 16 * (v >> 2) + 8 * hi;
      a.u[v] = *(const unsigned*)&As[buf * ASZ + (w * 16 + li) * AS_STRIDE + base];
    }
    #pragma unroll
    for (int ct = 0; ct < 4; ++ct) {
      AF bfr;
      #pragma unroll
      for (int v = 0; v < 8; ++v)
        bfr.u[v] = Ws[buf * WSZ + ((v & 3) + 4 * hi + 8 * (v >> 2)) * WS_STRIDE + ct * 16 + li];
      acc[ct] = __builtin_amdgcn_wmma_f32_16x16x32_f16(
          false, a.v, false, bfr.v, (short)0, acc[ct], false, false);
    }
  }

  // spill sim tile (+bias) to LDS for cross-wave softmax over K
  __syncthreads();
  #pragma unroll
  for (int ct = 0; ct < 4; ++ct) {
    float bv = bias[n0 + ct * 16 + li];
    #pragma unroll
    for (int r = 0; r < 8; ++r)
      simLds[(w * 16 + r + 8 * hi) * SIM_STRIDE + ct * 16 + li] = acc[ct][r] + bv;
  }
  __syncthreads();

  int g = tid >> 6;              // group 0..3 (one (b,n), 32 rows each)
  int c = tid & 63;
  float sv[KNN];
  float mx = -3.0e38f;
  #pragma unroll
  for (int k = 0; k < KNN; ++k) {
    sv[k] = simLds[(g * KNN + k) * SIM_STRIDE + c];
    mx = fmaxf(mx, sv[k]);
  }
  float sum = 0.0f;
  #pragma unroll
  for (int k = 0; k < KNN; ++k) { sv[k] = expf(sv[k] - mx); sum += sv[k]; }
  float inv = 1.0f / sum;
  size_t tok0 = (size_t)m0 + g * KNN;
  float av = 0.0f;
  #pragma unroll
  for (int k = 0; k < KNN; ++k)
    av += sv[k] * (float)Vpj[(tok0 + k) * 256 + n0 + c];
  Xout[((size_t)(m0 >> 5) + g) * 256 + n0 + c] = av * inv;
}

// ---------------------------------------------------------------------------
// Head: mean over N, LayerNorm, 256->512 GELU, 512->300.  One block / batch.
// ---------------------------------------------------------------------------
__global__ __launch_bounds__(256) void head_k(
    const float* __restrict__ X, const float* __restrict__ lg,
    const float* __restrict__ lb, const float* __restrict__ w1,
    const float* __restrict__ b1, const float* __restrict__ w2,
    const float* __restrict__ b2, float* __restrict__ out) {
  __shared__ float red[256];
  __shared__ float xb[256];
  __shared__ float hh[512];
  int b = blockIdx.x, t = threadIdx.x;
  float s = 0.0f;
  for (int n = 0; n < NPTS; ++n) s += X[((size_t)b * NPTS + n) * 256 + t];
  float xv = s * (1.0f / NPTS);
  red[t] = xv; __syncthreads();
  for (int st = 128; st > 0; st >>= 1) { if (t < st) red[t] += red[t + st]; __syncthreads(); }
  float mean = red[0] * (1.0f / 256.0f);
  __syncthreads();
  float dv = xv - mean;
  red[t] = dv * dv; __syncthreads();
  for (int st = 128; st > 0; st >>= 1) { if (t < st) red[t] += red[t + st]; __syncthreads(); }
  float var = red[0] * (1.0f / 256.0f);
  xb[t] = dv * rsqrtf(var + 1e-5f) * lg[t] + lb[t];
  __syncthreads();
  for (int rep = 0; rep < 2; ++rep) {
    int j = rep * 256 + t;
    float a = b1[j];
    for (int d = 0; d < 256; ++d) a += xb[d] * w1[(size_t)d * 512 + j];
    hh[j] = gelu_erf(a);
  }
  __syncthreads();
  for (int o = t; o < 300; o += 256) {
    float a = b2[o];
    for (int j = 0; j < 512; ++j) a += hh[j] * w2[(size_t)j * 300 + o];
    out[b * 300 + o] = a;
  }
}

// ---------------------------------------------------------------------------
extern "C" void kernel_launch(void* const* d_in, const int* in_sizes, int n_in,
                              void* d_out, int out_size, void* d_ws, size_t ws_size,
                              hipStream_t stream) {
  (void)in_sizes; (void)n_in; (void)out_size; (void)ws_size;
  const float* xyz     = (const float*)d_in[0];
  const float* emb_w   = (const float*)d_in[1];
  const float* emb_b   = (const float*)d_in[2];
  const float* eln_g   = (const float*)d_in[3];
  const float* eln_b   = (const float*)d_in[4];
  const float* qkv_w   = (const float*)d_in[5];
  const float* pos_w1  = (const float*)d_in[6];
  const float* pos_b1  = (const float*)d_in[7];
  const float* pos_w2  = (const float*)d_in[8];
  const float* pos_b2  = (const float*)d_in[9];
  const float* attn_w1 = (const float*)d_in[10];
  const float* attn_b1 = (const float*)d_in[11];
  const float* attn_w2 = (const float*)d_in[12];
  const float* attn_b2 = (const float*)d_in[13];
  const float* hln_g   = (const float*)d_in[14];
  const float* hln_b   = (const float*)d_in[15];
  const float* hw1     = (const float*)d_in[16];
  const float* hb1     = (const float*)d_in[17];
  const float* hw2     = (const float*)d_in[18];
  const float* hb2     = (const float*)d_in[19];
  float* out = (float*)d_out;

  // ---- workspace carve (256B aligned). ~515 MB, dominated by h. ----
  char* wsp = (char*)d_ws;
  auto carve = [&](size_t bytes) -> void* {
    void* p = (void*)wsp;
    wsp += (bytes + 255) & ~(size_t)255;
    return p;
  };
  unsigned* wqp  = (unsigned*)carve((size_t)4 * 128 * 768 * 4);   // packed qkv_w
  unsigned* w1p  = (unsigned*)carve((size_t)4 * 128 * 1024 * 4);  // packed attn_w1
  unsigned* w2p  = (unsigned*)carve((size_t)4 * 512 * 256 * 4);   // packed attn_w2
  float*    x0   = (float*)carve((size_t)ROWS * 256 * 4);
  float*    x1   = (float*)carve((size_t)ROWS * 256 * 4);
  float*    qkv  = (float*)carve((size_t)ROWS * 768 * 4);
  int*      idx  = (int*)carve((size_t)ROWS * KNN * 4);
  float*    rel  = (float*)carve((size_t)ROWS * KNN * 3 * 4);
  _Float16* ain  = (_Float16*)carve((size_t)TOK * 256 * 2);
  _Float16* vpj  = (_Float16*)carve((size_t)TOK * 256 * 2);
  _Float16* hbuf = (_Float16*)carve((size_t)TOK * 1024 * 2);

  // ---- weights -> f16 pair-packed LDS image (L2-resident thereafter) ----
  pack_w_k<<<((4 * 128 * 768) + 255) / 256, 256, 0, stream>>>(qkv_w, wqp, 4 * 256, 768);
  pack_w_k<<<((4 * 128 * 1024) + 255) / 256, 256, 0, stream>>>(attn_w1, w1p, 4 * 256, 1024);
  pack_w_k<<<((4 * 512 * 256) + 255) / 256, 256, 0, stream>>>(attn_w2, w2p, 4 * 1024, 256);

  // ---- embed + kNN (shared across layers) ----
  embed_k<<<ROWS, 256, 0, stream>>>(xyz, emb_w, emb_b, eln_g, eln_b, x0);
  knn_k<<<(ROWS + KNN_T - 1) / KNN_T, KNN_T, 0, stream>>>(xyz, idx, rel);

  float* xc = x0;
  float* xn = x1;
  for (int l = 0; l < 4; ++l) {
    // qkv = x @ Wqkv  (M=4900, N=768, K=256); A f32 (sync staged), W via TDM
    gemm_wmma_k<false, false, false, false, false>
        <<<dim3((ROWS + 127) / 128, 768 / 64), 256, 0, stream>>>(
            xc, wqp + (size_t)l * 128 * 768, nullptr, qkv, ROWS, 768, 256);
    // a_in / vpj (f16) with inlined pos-MLP
    build_tokens_k<<<ROWS, 256, 0, stream>>>(
        qkv, idx, rel, pos_w1 + (size_t)l * 3 * 64, pos_b1 + (size_t)l * 64,
        pos_w2 + (size_t)l * 64 * 256, pos_b2 + (size_t)l * 256, ain, vpj);
    // h = ReLU(a_in @ W1 + b1)  (M=156800, N=1024, K=256); full TDM staging
    gemm_wmma_k<true, true, true, true, true>
        <<<dim3(TOK / 128, 1024 / 64), 256, 0, stream>>>(
            ain, w1p + (size_t)l * 128 * 1024, attn_b1 + (size_t)l * 1024,
            hbuf, TOK, 1024, 256);
    // x_next = softmax_K(h @ W2 + b2) * vpj, fused (M=156800, N=256, K=1024)
    attn2_softmax_k<<<dim3(TOK / 128, 256 / 64), 256, 0, stream>>>(
        hbuf, w2p + (size_t)l * 512 * 256, attn_b2 + (size_t)l * 256,
        vpj, xn, 1024);
    float* tmp = xc; xc = xn; xn = tmp;
  }

  head_k<<<BATCH, 256, 0, stream>>>(xc, hln_g, hln_b, hw1, hb1, hw2, hb2, out);
}
